// Block_17763984736761
// MI455X (gfx1250) — compile-verified, matrix operand order checked by reading, not run on hardware
//
#include <hip/hip_runtime.h>
#include <math.h>

typedef __attribute__((ext_vector_type(16))) __bf16 v16bf;
typedef __attribute__((ext_vector_type(8)))  float  v8f;

#define B_   16
#define N_   1024
#define D_   256
#define H_   8
#define DH_  32
#define HID_ 1024
#define M_   (B_ * N_)   // 16384 tokens

// ---- CDNA5 async global->LDS path (GLOBAL_LOAD_ASYNC_TO_LDS_B128, ASYNCcnt) ----
#if defined(__has_builtin)
#if __has_builtin(__builtin_amdgcn_global_load_async_to_lds_b128) && \
    __has_builtin(__builtin_amdgcn_s_wait_asynccnt)
#define HAVE_ASYNC_LDS 1
#endif
#endif
#ifndef HAVE_ASYNC_LDS
#define HAVE_ASYNC_LDS 0
#endif

typedef int v4i32 __attribute__((vector_size(16)));
typedef __attribute__((address_space(1))) v4i32 gv4i;   // global
typedef __attribute__((address_space(3))) v4i32 lv4i;   // LDS

__device__ __forceinline__ void async_ld16(void* dst_lds, const void* src_global) {
#if HAVE_ASYNC_LDS
    __builtin_amdgcn_global_load_async_to_lds_b128(
        (gv4i*)(void*)src_global,
        (lv4i*)dst_lds,
        0, 0);
#else
    *(int4*)dst_lds = *(const int4*)src_global;   // sync fallback (vgpr bounce)
#endif
}

#if HAVE_ASYNC_LDS
#define ASYNC_WAIT(n) __builtin_amdgcn_s_wait_asynccnt(n)
#else
#define ASYNC_WAIT(n) do {} while (0)
#endif

// ---------------------------------------------------------------------------
// fp32 -> bf16 conversion (weights)
// ---------------------------------------------------------------------------
__global__ void f32_to_bf16(const float* __restrict__ s, __bf16* __restrict__ d, int n) {
    int i = blockIdx.x * 256 + threadIdx.x;
    if (i < n) d[i] = (__bf16)s[i];
}

// ---------------------------------------------------------------------------
// Row LayerNorm: fp32 [M,256] -> bf16 [M,256]. One 256-thread block per row.
// ---------------------------------------------------------------------------
__global__ __launch_bounds__(256) void ln_bf16(const float* __restrict__ x,
                                               const float* __restrict__ gam,
                                               const float* __restrict__ bet,
                                               __bf16* __restrict__ y) {
    __shared__ float red[256];
    const int row = blockIdx.x;
    const int t   = threadIdx.x;
    float v = x[(size_t)row * D_ + t];
    red[t] = v;
    __syncthreads();
    for (int s = 128; s > 0; s >>= 1) { if (t < s) red[t] += red[t + s]; __syncthreads(); }
    float mu = red[0] * (1.0f / D_);
    __syncthreads();
    float d = v - mu;
    red[t] = d * d;
    __syncthreads();
    for (int s = 128; s > 0; s >>= 1) { if (t < s) red[t] += red[t + s]; __syncthreads(); }
    float r = rsqrtf(red[0] * (1.0f / D_) + 1e-5f);
    y[(size_t)row * D_ + t] = (__bf16)(d * r * gam[t] + bet[t]);
}

// ---------------------------------------------------------------------------
// LDS-tiled bf16 WMMA GEMM: out[M,Nout] = A[M,K] * W[Nout,K]^T (+bias)(+gelu)(+res)
// Block tile 128x64, K-step 32, double-buffered async global->LDS staging.
// 8 waves / block; wave w computes rows [w*16, w*16+16) x all 64 cols:
// one A fragment reused across 4 WMMAs (4 accumulators) per K-step.
// ---------------------------------------------------------------------------
__global__ __launch_bounds__(256) void gemm_bf16_wmma(
    const __bf16* __restrict__ A, const __bf16* __restrict__ W,
    const float* __restrict__ bias, const float* __restrict__ residual,
    float* __restrict__ outF, __bf16* __restrict__ outB,
    int M, int Nout, int K, int do_gelu)
{
    __shared__ __bf16 sA[2][128 * 32];   // 8 KB per buffer
    __shared__ __bf16 sW[2][64 * 32];    // 4 KB per buffer

    const int t    = threadIdx.x;
    const int lane = t & 31;
    const int wv   = t >> 5;
    const int l15  = lane & 15;
    const int hl   = lane >> 4;

    const int ntiles = Nout >> 6;                 // Nout/64
    const int tm0 = (blockIdx.x / ntiles) * 128;
    const int tn0 = (blockIdx.x % ntiles) * 64;

    // stage(buf, k0): async-copy A[tm0..+128, k0..+32] and W[tn0..+64, k0..+32]
    // 3 x b128 async ops per thread per stage.
    auto stage = [&](int buf, int k0) {
#pragma unroll
        for (int i = 0; i < 2; ++i) {             // A tile: 512 16B chunks
            const int lin = t + i * 256;
            const int row = lin >> 2;
            const int seg = lin & 3;
            async_ld16(&sA[buf][row * 32 + seg * 8],
                       A + (size_t)(tm0 + row) * K + k0 + seg * 8);
        }
        {                                          // W tile: 256 16B chunks
            const int row = t >> 2;
            const int seg = t & 3;
            async_ld16(&sW[buf][row * 32 + seg * 8],
                       W + (size_t)(tn0 + row) * K + k0 + seg * 8);
        }
    };

    v8f acc[4] = {{}, {}, {}, {}};
    const int KT = K >> 5;

    stage(0, 0);
    for (int kt = 0; kt < KT; ++kt) {
        const int buf = kt & 1;
        if (kt + 1 < KT) {
            stage(buf ^ 1, (kt + 1) << 5);
            ASYNC_WAIT(3);    // next stage (3 ops) may stay in flight; kt's landed
        } else {
            ASYNC_WAIT(0);
        }
        __syncthreads();

        const v16bf a = *(const v16bf*)&sA[buf][(wv * 16 + l15) * 32 + hl * 16];
#pragma unroll
        for (int nt = 0; nt < 4; ++nt) {
            const v16bf b = *(const v16bf*)&sW[buf][(nt * 16 + l15) * 32 + hl * 16];
            acc[nt] = __builtin_amdgcn_wmma_f32_16x16x32_bf16(false, a, false, b,
                                                              (short)0, acc[nt],
                                                              false, false);
        }
        __syncthreads();
    }

    // epilogue
    const int row0 = tm0 + wv * 16;
#pragma unroll
    for (int nt = 0; nt < 4; ++nt) {
        const int col = tn0 + nt * 16 + l15;
        const float bv = bias ? bias[col] : 0.0f;
#pragma unroll
        for (int i = 0; i < 8; ++i) {
            const int row = row0 + i + 8 * hl;    // C/D layout: vgpr i, lane-half
            float v = acc[nt][i] + bv;
            if (do_gelu) v = 0.5f * v * (1.0f + erff(v * 0.70710678118654752f));
            const size_t idx = (size_t)row * Nout + col;
            if (residual) v += residual[idx];
            if (outF) outF[idx] = v;
            if (outB) outB[idx] = (__bf16)v;
        }
    }
}

// ---------------------------------------------------------------------------
// Attention: one wave per (b, h, 16-query-row tile). DH=32 => single K-step
// WMMA for scores; 16x1024 fp32 score strip staged in 64KB LDS (WGP has 320KB).
// qkv layout: bf16 [M, 768] = [q(256) | k(256) | v(256)], head h at h*32.
// ---------------------------------------------------------------------------
__global__ __launch_bounds__(32) void attn_wmma(const __bf16* __restrict__ qkv,
                                                const float* __restrict__ scale_arr,
                                                __bf16* __restrict__ o)
{
    __shared__ float sS[16 * N_];               // 64 KB
    const int lane = threadIdx.x & 31;
    const int qt = blockIdx.x & 63;             // N/16 = 64 query tiles
    const int bh = blockIdx.x >> 6;
    const int b  = bh >> 3;                     // H = 8
    const int h  = bh & 7;
    const int l15 = lane & 15;
    const int hl  = lane >> 4;
    const float sc = scale_arr[h];

    const size_t rs = 3 * D_;                   // 768 elements per token row
    const __bf16* qbase = qkv + (size_t)b * N_ * rs + (size_t)h * DH_;
    const __bf16* kbase = qbase + D_;
    const __bf16* vbase = qbase + 2 * D_;

    // Q fragment: row = qt*16 + l15, K chunk = hl*16 (DH=32 exactly one WMMA K)
    const v16bf qa = *(const v16bf*)(qbase + (size_t)(qt * 16 + l15) * rs + hl * 16);

    // ---- scores S = (Q K^T) * scale, diagonal masked to -inf ----
    for (int jt = 0; jt < 64; ++jt) {
        v16bf kb = *(const v16bf*)(kbase + (size_t)(jt * 16 + l15) * rs + hl * 16);
        v8f s = {};
        s = __builtin_amdgcn_wmma_f32_16x16x32_bf16(false, qa, false, kb,
                                                    (short)0, s, false, false);
        const int col = jt * 16 + l15;
#pragma unroll
        for (int i = 0; i < 8; ++i) {
            const int r  = i + 8 * hl;
            const int gi = qt * 16 + r;
            sS[r * N_ + col] = (gi == col) ? -INFINITY : s[i] * sc;
        }
    }
    __syncthreads();

    // ---- softmax: row r = l15, each lane-half scans 512 entries ----
    {
        const int r = l15;
        float mx = -INFINITY;
        for (int j = hl * 512; j < hl * 512 + 512; ++j) mx = fmaxf(mx, sS[r * N_ + j]);
        mx = fmaxf(mx, __shfl_xor(mx, 16, 32));
        float sum = 0.0f;
        for (int j = hl * 512; j < hl * 512 + 512; ++j) sum += __expf(sS[r * N_ + j] - mx);
        sum += __shfl_xor(sum, 16, 32);
        const float inv = 1.0f / sum;
        for (int j = hl * 512; j < hl * 512 + 512; ++j)
            sS[r * N_ + j] = __expf(sS[r * N_ + j] - mx) * inv;
    }
    __syncthreads();

    // ---- O = P @ V : P frag 16x32 from LDS, V frags column-major gathers ----
    v8f o0 = {}, o1 = {};
    for (int kt = 0; kt < 32; ++kt) {
        v16bf pa;
#pragma unroll
        for (int tt = 0; tt < 16; ++tt)
            pa[tt] = (__bf16)sS[l15 * N_ + kt * 32 + hl * 16 + tt];
        const __bf16* vk = vbase + (size_t)(kt * 32 + hl * 16) * rs;
        v16bf vb0, vb1;
#pragma unroll
        for (int tt = 0; tt < 16; ++tt) {       // lane = output dh column
            vb0[tt] = vk[(size_t)tt * rs + l15];
            vb1[tt] = vk[(size_t)tt * rs + 16 + l15];
        }
        o0 = __builtin_amdgcn_wmma_f32_16x16x32_bf16(false, pa, false, vb0,
                                                     (short)0, o0, false, false);
        o1 = __builtin_amdgcn_wmma_f32_16x16x32_bf16(false, pa, false, vb1,
                                                     (short)0, o1, false, false);
    }

    // store O as bf16 into [M, 256] at column h*32
    __bf16* obase = o + ((size_t)b * N_ + qt * 16) * D_ + h * DH_;
#pragma unroll
    for (int i = 0; i < 8; ++i) {
        const int row = i + 8 * hl;
        obase[(size_t)row * D_ + l15]      = (__bf16)o0[i];
        obase[(size_t)row * D_ + 16 + l15] = (__bf16)o1[i];
    }
}

// ---------------------------------------------------------------------------
// Host-side launch sequence
// ---------------------------------------------------------------------------
extern "C" void kernel_launch(void* const* d_in, const int* in_sizes, int n_in,
                              void* d_out, int out_size, void* d_ws, size_t ws_size,
                              hipStream_t stream)
{
    const float* x      = (const float*)d_in[0];
    const float* ln1_g  = (const float*)d_in[1];
    const float* ln1_b  = (const float*)d_in[2];
    const float* qkv_w  = (const float*)d_in[3];
    const float* scale  = (const float*)d_in[4];
    const float* proj_w = (const float*)d_in[5];
    const float* proj_b = (const float*)d_in[6];
    const float* ln2_g  = (const float*)d_in[7];
    const float* ln2_b  = (const float*)d_in[8];
    const float* fc1_w  = (const float*)d_in[9];
    const float* fc1_b  = (const float*)d_in[10];
    const float* fc2_w  = (const float*)d_in[11];
    const float* fc2_b  = (const float*)d_in[12];
    float* out = (float*)d_out;

    char* ws = (char*)d_ws;
    size_t off = 0;
    auto alloc = [&](size_t bytes) -> void* {
        void* p = ws + off;
        off += (bytes + 255) & ~(size_t)255;
        return p;
    };

    __bf16* qkv_w_bf  = (__bf16*)alloc((size_t)3 * D_ * D_ * 2);
    __bf16* proj_w_bf = (__bf16*)alloc((size_t)D_ * D_ * 2);
    __bf16* fc1_w_bf  = (__bf16*)alloc((size_t)HID_ * D_ * 2);
    __bf16* fc2_w_bf  = (__bf16*)alloc((size_t)D_ * HID_ * 2);
    __bf16* hA        = (__bf16*)alloc((size_t)M_ * D_ * 2);        // ln1(x)
    __bf16* qkvb      = (__bf16*)alloc((size_t)M_ * 3 * D_ * 2);    // qkv proj
    __bf16* ob        = (__bf16*)alloc((size_t)M_ * D_ * 2);        // attn out
    float*  x1        = (float*)alloc((size_t)M_ * D_ * 4);         // x + proj
    __bf16* h2        = (__bf16*)alloc((size_t)M_ * D_ * 2);        // ln2(x1)
    __bf16* a1        = (__bf16*)alloc((size_t)M_ * HID_ * 2);      // gelu(fc1)

    // weights -> bf16
    f32_to_bf16<<<(3 * D_ * D_ + 255) / 256, 256, 0, stream>>>(qkv_w,  qkv_w_bf,  3 * D_ * D_);
    f32_to_bf16<<<(D_ * D_ + 255) / 256,     256, 0, stream>>>(proj_w, proj_w_bf, D_ * D_);
    f32_to_bf16<<<(HID_ * D_ + 255) / 256,   256, 0, stream>>>(fc1_w,  fc1_w_bf,  HID_ * D_);
    f32_to_bf16<<<(D_ * HID_ + 255) / 256,   256, 0, stream>>>(fc2_w,  fc2_w_bf,  D_ * HID_);

    // --- attention branch ---
    ln_bf16<<<M_, 256, 0, stream>>>(x, ln1_g, ln1_b, hA);
    gemm_bf16_wmma<<<(M_ / 128) * (3 * D_ / 64), 256, 0, stream>>>(
        hA, qkv_w_bf, nullptr, nullptr, nullptr, qkvb, M_, 3 * D_, D_, 0);
    attn_wmma<<<B_ * H_ * (N_ / 16), 32, 0, stream>>>(qkvb, scale, ob);
    gemm_bf16_wmma<<<(M_ / 128) * (D_ / 64), 256, 0, stream>>>(
        ob, proj_w_bf, proj_b, x, x1, nullptr, M_, D_, D_, 0);

    // --- MLP branch ---
    ln_bf16<<<M_, 256, 0, stream>>>(x1, ln2_g, ln2_b, h2);
    gemm_bf16_wmma<<<(M_ / 128) * (HID_ / 64), 256, 0, stream>>>(
        h2, fc1_w_bf, fc1_b, nullptr, nullptr, a1, M_, HID_, D_, 1);
    gemm_bf16_wmma<<<(M_ / 128) * (D_ / 64), 256, 0, stream>>>(
        a1, fc2_w_bf, fc2_b, x1, out, nullptr, M_, D_, HID_, 0);
}
